// RingDilatedAttentionV2_86517821216466
// MI455X (gfx1250) — matrix operand order repaired and checked by reading, not compiled, not run on hardware
//
#include <hip/hip_runtime.h>
#include <cmath>

typedef __attribute__((ext_vector_type(16))) _Float16 v16h;
typedef __attribute__((ext_vector_type(8)))  _Float16 v8h;
typedef __attribute__((ext_vector_type(8)))  float    v8f;

#define SEQ 8192
#define NH  8
#define DH  64
#define ROWSTRIDE (NH*DH)   // 512 floats per sequence position
#define KV  64              // keys per streamed chunk
#define NWAVE 8             // waves per block (wave32)

__device__ __forceinline__ v16h cat16(v8h a, v8h b) {
  v16h r;
#pragma unroll
  for (int i = 0; i < 8; ++i) { r[i] = a[i]; r[i + 8] = b[i]; }
  return r;
}

// Butterfly reduce over each 16-lane row via DPP16 (pure VALU, no LDS).
// Controls: quad_perm xor1 = 0xB1, quad_perm xor2 = 0x4E,
//           row_half_mirror (xor7) = 0x141, row_mirror (xor15) = 0x140.
// span{1,2,7,15} = full 16-lane group.
__device__ __forceinline__ float dpp_row_max(float v) {
  int p;
  p = __builtin_amdgcn_update_dpp(0, __float_as_int(v), 0xB1,  0xF, 0xF, true);
  v = fmaxf(v, __int_as_float(p));
  p = __builtin_amdgcn_update_dpp(0, __float_as_int(v), 0x4E,  0xF, 0xF, true);
  v = fmaxf(v, __int_as_float(p));
  p = __builtin_amdgcn_update_dpp(0, __float_as_int(v), 0x141, 0xF, 0xF, true);
  v = fmaxf(v, __int_as_float(p));
  p = __builtin_amdgcn_update_dpp(0, __float_as_int(v), 0x140, 0xF, 0xF, true);
  v = fmaxf(v, __int_as_float(p));
  return v;
}

__device__ __forceinline__ float dpp_row_sum(float v) {
  int p;
  p = __builtin_amdgcn_update_dpp(0, __float_as_int(v), 0xB1,  0xF, 0xF, true);
  v += __int_as_float(p);
  p = __builtin_amdgcn_update_dpp(0, __float_as_int(v), 0x4E,  0xF, 0xF, true);
  v += __int_as_float(p);
  p = __builtin_amdgcn_update_dpp(0, __float_as_int(v), 0x141, 0xF, 0xF, true);
  v += __int_as_float(p);
  p = __builtin_amdgcn_update_dpp(0, __float_as_int(v), 0x140, 0xF, 0xF, true);
  v += __int_as_float(p);
  return v;
}

__global__ __launch_bounds__(256)
void ring_attn_fwd(const float* __restrict__ Qg,
                   const float* __restrict__ Kg,
                   const float* __restrict__ Vg,
                   float* __restrict__ Og)
{
  // K chunk row-major [key][d]; V chunk transposed [d][key]; padded rows
  // (72 halves = 144B, multiple of 16B) keep v8h casts aligned and stagger banks.
  __shared__ alignas(64) _Float16 sK [KV][72];
  __shared__ alignas(64) _Float16 sVt[DH][72];
  __shared__ alignas(64) _Float16 sP [NWAVE][16][KV];   // per-wave P scratch

  const int tid  = threadIdx.x;
  const int wave = tid >> 5;
  const int lane = tid & 31;
  const int lo   = lane & 15;
  const int hi   = lane >> 4;          // half-wave select (wave32)
  const int head = blockIdx.x >> 6;    // 8 heads
  const int qblk = blockIdx.x & 63;    // 64 blocks of 128 queries
  const int qbase = qblk * (NWAVE * 16) + wave * 16;

  // ---- Q tile -> two 16x32 A fragments, pre-scaled by 1/sqrt(64) ----
  // A layout: row m = lo; elem e -> K=(hi?16:0)+e
  v16h aQ[2];
  {
    const float* qrow = Qg + (size_t)(qbase + lo) * ROWSTRIDE + head * DH;
#pragma unroll
    for (int f = 0; f < 2; ++f) {
      const int d0 = f * 32 + (hi ? 16 : 0);
      const int d1 = f * 32 + (hi ? 24 : 8);
#pragma unroll
      for (int e = 0; e < 8; ++e) aQ[f][e]     = (_Float16)(qrow[d0 + e] * 0.125f);
#pragma unroll
      for (int e = 0; e < 8; ++e) aQ[f][e + 8] = (_Float16)(qrow[d1 + e] * 0.125f);
    }
  }

  float mrow[8], lrow[8];
  v8f acc[4];
  {
    v8f z = {};
#pragma unroll
    for (int t = 0; t < 4; ++t) acc[t] = z;
#pragma unroll
    for (int r = 0; r < 8; ++r) { mrow[r] = -1e30f; lrow[r] = 0.f; }
  }

  for (int j = 0; j < SEQ; j += KV) {
    __syncthreads();
    // ---- cooperative stage of K & V chunk (64 keys x 64 dims, fp32->f16) ----
#pragma unroll
    for (int p = 0; p < 4; ++p) {
      const int l4  = p * 256 + tid;      // float4 slot 0..1023
      const int key = l4 >> 4;            // 16 float4 per key row
      const int d   = (l4 & 15) * 4;
      const size_t g = (size_t)(j + key) * ROWSTRIDE + head * DH + d;
      const float4 kk = *(const float4*)(Kg + g);
      const float4 vv = *(const float4*)(Vg + g);
      sK[key][d + 0] = (_Float16)kk.x; sK[key][d + 1] = (_Float16)kk.y;
      sK[key][d + 2] = (_Float16)kk.z; sK[key][d + 3] = (_Float16)kk.w;
      sVt[d + 0][key] = (_Float16)vv.x; sVt[d + 1][key] = (_Float16)vv.y;
      sVt[d + 2][key] = (_Float16)vv.z; sVt[d + 3][key] = (_Float16)vv.w;
    }
    __syncthreads();

    // ---- S = (Q*scale) K^T : four 16x16 tiles covering 64 keys ----
    v8f S[4];
#pragma unroll
    for (int nt = 0; nt < 4; ++nt) {
      const int key = nt * 16 + lo;       // B layout: col n = lo; elem e -> K=(hi?16:0)+e
      v16h b0 = cat16(*(const v8h*)&sK[key][hi * 16 + 0],
                      *(const v8h*)&sK[key][hi * 16 + 8]);        // d 0..31
      v16h b1 = cat16(*(const v8h*)&sK[key][32 + hi * 16 + 0],
                      *(const v8h*)&sK[key][32 + hi * 16 + 8]);   // d 32..63
      v8f c = {};
      c = __builtin_amdgcn_wmma_f32_16x16x32_f16(false, aQ[0], false, b0, (short)0, c, false, false);
      c = __builtin_amdgcn_wmma_f32_16x16x32_f16(false, aQ[1], false, b1, (short)0, c, false, false);
      S[nt] = c;
    }

    // ---- online softmax (C layout: row = r + 8*hi, col = lo) ----
    float corr[8];
#pragma unroll
    for (int r = 0; r < 8; ++r) {
      float cm = fmaxf(fmaxf(S[0][r], S[1][r]), fmaxf(S[2][r], S[3][r]));
      cm = dpp_row_max(cm);
      const float mn = fmaxf(mrow[r], cm);
      corr[r] = __expf(mrow[r] - mn);
      mrow[r] = mn;
    }
#pragma unroll
    for (int nt = 0; nt < 4; ++nt)
#pragma unroll
      for (int r = 0; r < 8; ++r)
        S[nt][r] = __expf(S[nt][r] - mrow[r]);
#pragma unroll
    for (int r = 0; r < 8; ++r) {
      float rs = (S[0][r] + S[1][r]) + (S[2][r] + S[3][r]);
      rs = dpp_row_sum(rs);
      lrow[r] = lrow[r] * corr[r] + rs;
    }
#pragma unroll
    for (int t = 0; t < 4; ++t)
#pragma unroll
      for (int r = 0; r < 8; ++r) acc[t][r] *= corr[r];

    // ---- P: C layout -> A layout via per-wave LDS round-trip ----
#pragma unroll
    for (int nt = 0; nt < 4; ++nt)
#pragma unroll
      for (int r = 0; r < 8; ++r)
        sP[wave][r + hi * 8][nt * 16 + lo] = (_Float16)S[nt][r];
    asm volatile("" ::: "memory");      // keep compiler from reordering; DS ops are in-order per wave
    // Two A fragments of P: keys 0..31 and 32..63
    v16h aP[2];
#pragma unroll
    for (int f = 0; f < 2; ++f)
      aP[f] = cat16(*(const v8h*)&sP[wave][lo][f * 32 + (hi ? 16 : 0)],
                    *(const v8h*)&sP[wave][lo][f * 32 + (hi ? 24 : 8)]);

    // ---- O += P V  (contract 64 keys; 4 output d-tiles) ----
#pragma unroll
    for (int nt = 0; nt < 4; ++nt) {
      const int d = nt * 16 + lo;         // B layout: col n = lo -> d column
      v16h bv0 = cat16(*(const v8h*)&sVt[d][hi * 16 + 0],
                       *(const v8h*)&sVt[d][hi * 16 + 8]);        // keys 0..31
      v16h bv1 = cat16(*(const v8h*)&sVt[d][32 + hi * 16 + 0],
                       *(const v8h*)&sVt[d][32 + hi * 16 + 8]);   // keys 32..63
      acc[nt] = __builtin_amdgcn_wmma_f32_16x16x32_f16(false, aP[0], false, bv0, (short)0, acc[nt], false, false);
      acc[nt] = __builtin_amdgcn_wmma_f32_16x16x32_f16(false, aP[1], false, bv1, (short)0, acc[nt], false, false);
    }
  }

  // ---- epilogue: O / l, scatter to [n, h, d] fp32 ----
#pragma unroll
  for (int r = 0; r < 8; ++r) {
    const float inv = 1.0f / lrow[r];
    const size_t q = (size_t)(qbase + r + hi * 8);
#pragma unroll
    for (int nt = 0; nt < 4; ++nt)
      Og[q * ROWSTRIDE + head * DH + nt * 16 + lo] = acc[nt][r] * inv;
  }
}

extern "C" void kernel_launch(void* const* d_in, const int* in_sizes, int n_in,
                              void* d_out, int out_size, void* d_ws, size_t ws_size,
                              hipStream_t stream) {
  const float* Q = (const float*)d_in[0];
  const float* K = (const float*)d_in[1];
  const float* V = (const float*)d_in[2];
  // d_in[3] = ring_size: online softmax makes the result independent of it.
  float* O = (float*)d_out;
  const int qblocks = SEQ / (NWAVE * 16);     // 64
  dim3 grid(NH * qblocks);                    // 512 blocks
  ring_attn_fwd<<<grid, 256, 0, stream>>>(Q, K, V, O);
}
